// ProSTformer_11081015623733
// MI455X (gfx1250) — matrix operand order repaired and checked by reading, compiled, not run on hardware
//
#include <hip/hip_runtime.h>
#include <hip/hip_bf16.h>
#include <math.h>

// ============================================================================
// ProSTformer forward for MI455X (gfx1250, wave32, WMMA).
//
// Precision strategy: fp32 master activations; all GEMMs run through
// v_wmma_f32_16x16x32_f16 (f16 A/B, f32 accumulate). ~0.93 TFLOP per forward,
// weights 67MB in f16 (L2-resident on the 192MB L2), HBM traffic negligible
// at 23.3 TB/s -> compute bound -> tensor-core GEMM everywhere.
// A-tile staging uses CDNA5 async global->LDS copies (ASYNCcnt).
// ============================================================================

typedef __attribute__((ext_vector_type(16))) _Float16 v16h;
typedef __attribute__((ext_vector_type(8)))  float    v8f;
typedef _Float16 h16;

#define BATCH 256
#define TOK   192          // P*F*S*N
#define GTOK  48           // P*F*S (dim-512 tokens)

// ---------------- GEMM: C = act(A[M,K](f16) x W[K,N](f16) + bias [+res]) ----
// Workgroup tile 128x64, K-step 64 (two 16x16x32 WMMA passes per barrier).
// 8 waves; wave w owns rows w*16..w*16+15 and all 4 N-subtiles.
#define BM 128
#define BN 64
#define BK 64
#define LDA 72   // halfs, padded row stride for A tile [128][64+8]
#define LDB 66   // halfs, padded stride for K-transposed B tile [64][64+2]

__device__ __forceinline__ void async_load_b128(unsigned lds_off, const void* gaddr)
{
    // GLOBAL_LOAD_ASYNC_TO_LDS_B128: per-lane 16B global -> LDS, ASYNCcnt.
    asm volatile("global_load_async_to_lds_b128 %0, %1, off"
                 :: "v"(lds_off), "v"(gaddr) : "memory");
}

__device__ __forceinline__ void wait_asynccnt0()
{
#if __has_builtin(__builtin_amdgcn_s_wait_asynccnt)
    __builtin_amdgcn_s_wait_asynccnt(0);
#else
    asm volatile("s_wait_asynccnt 0x0" ::: "memory");
#endif
}

__global__ __launch_bounds__(256)
void gemm_wmma_kernel(const h16* __restrict__ A, const h16* __restrict__ W,
                      const float* __restrict__ bias, const float* __restrict__ res,
                      float* __restrict__ C, h16* __restrict__ C16,
                      int M, int N, int K, int act_gelu)
{
    __shared__ __align__(16) h16 As[BM * LDA];
    __shared__ __align__(16) h16 Bs[BN * LDB];

    const int tid  = threadIdx.x;
    const int wave = tid >> 5;
    const int lane = tid & 31;
    const int laneM = lane & 15;     // column-of-16 within fragment
    const int hi    = lane >> 4;     // half-select (lanes 0-15 vs 16-31)
    const int m0 = blockIdx.y * BM;
    const int n0 = blockIdx.x * BN;

    v8f acc[4] = {};

    for (int k0 = 0; k0 < K; k0 += BK) {
        // ---- stage A tile (128x64 halfs) via async global->LDS (ASYNCcnt):
        // 32 halfs/thread = four b128 transfers, no VGPR round-trip.
        {
            const int row = tid >> 1;
            const int hc  = (tid & 1) * 32;
            const h16* gp = A + (size_t)(m0 + row) * K + k0 + hc;
            const unsigned lds0 =
                (unsigned)(unsigned long long)&As[row * LDA + hc];
            async_load_b128(lds0,       gp);
            async_load_b128(lds0 + 16u, gp + 8);
            async_load_b128(lds0 + 32u, gp + 16);
            async_load_b128(lds0 + 48u, gp + 24);
            if (k0 + BK < K)
                __builtin_prefetch(gp + BK, 0, 1);   // global_prefetch_b8
        }
        // ---- stage B tile (64x64 halfs), stored K-transposed Bs[n][k]
        // (element transpose scatter -> must go through VGPRs)
        {
            const int krow = tid >> 2;
            const int cg   = (tid & 3) * 16;
            const h16* gp = W + (size_t)(k0 + krow) * N + n0 + cg;
            union { float4 f; h16 h[8]; } t0, t1;
            t0.f = *reinterpret_cast<const float4*>(gp);
            t1.f = *reinterpret_cast<const float4*>(gp + 8);
            #pragma unroll
            for (int e = 0; e < 8; ++e) {
                Bs[(cg + e) * LDB + krow]     = t0.h[e];
                Bs[(cg + 8 + e) * LDB + krow] = t1.h[e];
            }
            if (k0 + BK < K)
                __builtin_prefetch(W + (size_t)(k0 + BK + krow) * N + n0 + cg, 0, 1);
        }
        wait_asynccnt0();       // A tile landed in LDS
        __syncthreads();        // + B tile visible workgroup-wide

        #pragma unroll
        for (int kk = 0; kk < 2; ++kk) {
            // ---- A fragment: contiguous K order matching B.
            // lane L: M = L%16; vgpr i holds K pair: K=2i(+16 if L>=16)
            union { v16h v; unsigned u[8]; } af;
            const int am = wave * 16 + laneM;
            #pragma unroll
            for (int i = 0; i < 8; ++i) {
                const int kx = 2 * i + hi * 16;
                af.u[i] = *reinterpret_cast<const unsigned*>(
                              &As[am * LDA + kk * 32 + kx]);
            }

            #pragma unroll
            for (int j = 0; j < 4; ++j) {
                // B fragment: lanes 0-15 K=0..15, lanes 16-31 K=16..31;
                // vgpr i halves = rows (kb+2i, kb+2i+1), col = laneM.
                union { v16h v; unsigned u[8]; } bf;
                const int bn = j * 16 + laneM;
                const int kb = kk * 32 + hi * 16;
                #pragma unroll
                for (int i = 0; i < 8; ++i)
                    bf.u[i] = *reinterpret_cast<const unsigned*>(
                                  &Bs[bn * LDB + kb + 2 * i]);

                acc[j] = __builtin_amdgcn_wmma_f32_16x16x32_f16(
                             false, af.v, false, bf.v, (short)0, acc[j], false, false);
            }
        }
        __syncthreads();
    }

    // ---- epilogue: C/D layout: vgpr r, lane L -> M = r + 8*(L>=16), N = L%16
    #pragma unroll
    for (int j = 0; j < 4; ++j) {
        #pragma unroll
        for (int r = 0; r < 8; ++r) {
            const int row = m0 + wave * 16 + r + hi * 8;
            const int col = n0 + j * 16 + laneM;
            float val = acc[j][r] + bias[col];
            if (res)      val += res[(size_t)row * N + col];
            if (act_gelu) val = 0.5f * val * (1.0f + erff(val * 0.70710678118f));
            if (C)   C[(size_t)row * N + col]   = val;
            if (C16) C16[(size_t)row * N + col] = (h16)val;
        }
    }
}

// ---------------- LayerNorm: one wave32 per row, f16 output -----------------
// pad>0 remaps output row r -> (r/group)*(group+pad) + pad + r%group
// (used to write LN(x_) into slots 1..48 of the 49-slot ta1 concat buffer).
__global__ void ln_kernel(const float* __restrict__ x, const float* __restrict__ g,
                          const float* __restrict__ b, h16* __restrict__ out,
                          int rows, int dm, int group, int pad)
{
    const int wid  = (blockIdx.x * blockDim.x + threadIdx.x) >> 5;
    const int lane = threadIdx.x & 31;
    if (wid >= rows) return;
    const float* xr = x + (size_t)wid * dm;

    float s = 0.f;
    for (int c = lane; c < dm; c += 32) s += xr[c];
    #pragma unroll
    for (int o = 16; o > 0; o >>= 1) s += __shfl_xor(s, o, 32);
    const float mean = s / dm;

    float v = 0.f;
    for (int c = lane; c < dm; c += 32) { float d = xr[c] - mean; v += d * d; }
    #pragma unroll
    for (int o = 16; o > 0; o >>= 1) v += __shfl_xor(v, o, 32);
    const float inv = rsqrtf(v / dm + 1e-6f);

    size_t orow = pad ? ((size_t)(wid / group) * (group + pad) + pad + wid % group)
                      : (size_t)wid;
    h16* op = out + orow * dm;
    for (int c = lane; c < dm; c += 32)
        op[c] = (h16)((xr[c] - mean) * inv * g[c] + b[c]);
}

// ---------------- attention kernels (tiny key counts -> VALU) ---------------
// sa: groups (p,f,s), 4 keys (n), H=4, DH=32, tokens (B*192,128)
__global__ void attn_sa_kernel(const float* __restrict__ q, const float* __restrict__ k,
                               const float* __restrict__ v, h16* __restrict__ out)
{
    const int idx = blockIdx.x * blockDim.x + threadIdx.x;
    if (idx >= BATCH * 4 * 48 * 4) return;
    const int n  = idx & 3;
    const int g  = (idx >> 2) % 48;
    const int hb = (idx >> 2) / 48;
    const int h  = hb & 3;
    const int b  = hb >> 2;
    const float scale = 0.17677669529f;   // 32^-0.5
    const size_t tb = (size_t)b * TOK + g * 4;
    const float* qp = q + (tb + n) * 128 + h * 32;
    float s[4], mx = -3.0e38f;
    for (int j = 0; j < 4; ++j) {
        const float* kp = k + (tb + j) * 128 + h * 32;
        float d = 0.f;
        for (int c = 0; c < 32; ++c) d += qp[c] * kp[c];
        s[j] = d * scale; mx = fmaxf(mx, s[j]);
    }
    float den = 0.f;
    for (int j = 0; j < 4; ++j) { s[j] = __expf(s[j] - mx); den += s[j]; }
    float o[32];
    for (int c = 0; c < 32; ++c) o[c] = 0.f;
    for (int j = 0; j < 4; ++j) {
        const float w = s[j] / den;
        const float* vp = v + (tb + j) * 128 + h * 32;
        for (int c = 0; c < 32; ++c) o[c] += w * vp[c];
    }
    h16* op = out + (tb + n) * 128 + h * 32;
    for (int c = 0; c < 32; ++c) op[c] = (h16)o[c];
}

// ta: groups (p,s,n), 4 keys (f, token stride 16), H=4, DH=32
__global__ void attn_ta_kernel(const float* __restrict__ q, const float* __restrict__ k,
                               const float* __restrict__ v, h16* __restrict__ out)
{
    const int idx = blockIdx.x * blockDim.x + threadIdx.x;
    if (idx >= BATCH * 4 * 3 * 4 * 4 * 4) return;
    const int f  = idx & 3;
    const int n  = (idx >> 2) & 3;
    const int sp = (idx >> 4) & 3;
    const int rest = idx >> 6;
    const int p  = rest % 3;
    const int hb = rest / 3;
    const int h  = hb & 3;
    const int b  = hb >> 2;
    const float scale = 0.17677669529f;
    const size_t base = (size_t)b * TOK + p * 64 + sp * 4 + n;  // + f*16
    const float* qp = q + (base + f * 16) * 128 + h * 32;
    float s[4], mx = -3.0e38f;
    for (int j = 0; j < 4; ++j) {
        const float* kp = k + (base + j * 16) * 128 + h * 32;
        float d = 0.f;
        for (int c = 0; c < 32; ++c) d += qp[c] * kp[c];
        s[j] = d * scale; mx = fmaxf(mx, s[j]);
    }
    float den = 0.f;
    for (int j = 0; j < 4; ++j) { s[j] = __expf(s[j] - mx); den += s[j]; }
    float o[32];
    for (int c = 0; c < 32; ++c) o[c] = 0.f;
    for (int j = 0; j < 4; ++j) {
        const float w = s[j] / den;
        const float* vp = v + (base + j * 16) * 128 + h * 32;
        for (int c = 0; c < 32; ++c) o[c] += w * vp[c];
    }
    h16* op = out + (base + f * 16) * 128 + h * 32;
    for (int c = 0; c < 32; ++c) op[c] = (h16)o[c];
}

// sa1: tokens (B*48,512), groups (p,f), 4 keys (s), H1=8, DH1=64
__global__ void attn_sa1_kernel(const float* __restrict__ q, const float* __restrict__ k,
                                const float* __restrict__ v, h16* __restrict__ out)
{
    const int idx = blockIdx.x * blockDim.x + threadIdx.x;
    if (idx >= BATCH * 8 * 12 * 4) return;
    const int sq = idx & 3;
    const int pf = (idx >> 2) % 12;
    const int hb = (idx >> 2) / 12;
    const int h  = hb & 7;
    const int b  = hb >> 3;
    const float scale = 0.125f;          // 64^-0.5
    const size_t tb = (size_t)b * GTOK + pf * 4;
    const float* qp = q + (tb + sq) * 512 + h * 64;
    float s[4], mx = -3.0e38f;
    for (int j = 0; j < 4; ++j) {
        const float* kp = k + (tb + j) * 512 + h * 64;
        float d = 0.f;
        for (int c = 0; c < 64; ++c) d += qp[c] * kp[c];
        s[j] = d * scale; mx = fmaxf(mx, s[j]);
    }
    float den = 0.f;
    for (int j = 0; j < 4; ++j) { s[j] = __expf(s[j] - mx); den += s[j]; }
    float o[64];
    for (int c = 0; c < 64; ++c) o[c] = 0.f;
    for (int j = 0; j < 4; ++j) {
        const float w = s[j] / den;
        const float* vp = v + (tb + j) * 512 + h * 64;
        for (int c = 0; c < 64; ++c) o[c] += w * vp[c];
    }
    h16* op = out + (tb + sq) * 512 + h * 64;
    for (int c = 0; c < 64; ++c) op[c] = (h16)o[c];
}

// ta1: cat tokens (B*49,512): slot0=ext, slot 1+pf*4+s = y. per group s:
// 13 slots [ext, pf=0..11]; ext output taken from group s==1, query slot 0.
__global__ void attn_ta1_kernel(const float* __restrict__ q, const float* __restrict__ k,
                                const float* __restrict__ v, h16* __restrict__ out)
{
    const int idx = blockIdx.x * blockDim.x + threadIdx.x;
    if (idx >= BATCH * 8 * 4 * 13) return;
    const int i  = idx % 13;
    const int r  = idx / 13;
    const int sp = r & 3;
    const int hb = r >> 2;
    const int h  = hb & 7;
    const int b  = hb >> 3;
    const float scale = 0.125f;
    const size_t cb = (size_t)b * 49;
    auto rowOf = [&](int j) -> size_t {
        return (j == 0) ? cb : cb + 1 + (size_t)(j - 1) * 4 + sp;
    };
    const float* qp = q + rowOf(i) * 512 + h * 64;
    float s[13], mx = -3.0e38f;
    for (int j = 0; j < 13; ++j) {
        const float* kp = k + rowOf(j) * 512 + h * 64;
        float d = 0.f;
        for (int c = 0; c < 64; ++c) d += qp[c] * kp[c];
        s[j] = d * scale; mx = fmaxf(mx, s[j]);
    }
    float den = 0.f;
    for (int j = 0; j < 13; ++j) { s[j] = __expf(s[j] - mx); den += s[j]; }
    float o[64];
    for (int c = 0; c < 64; ++c) o[c] = 0.f;
    for (int j = 0; j < 13; ++j) {
        const float w = s[j] / den;
        const float* vp = v + rowOf(j) * 512 + h * 64;
        for (int c = 0; c < 64; ++c) o[c] += w * vp[c];
    }
    if (i == 0) {
        if (sp == 1) {                       // ext result comes from s-slot 1
            h16* op = out + cb * 512 + h * 64;
            for (int c = 0; c < 64; ++c) op[c] = (h16)o[c];
        }
    } else {
        h16* op = out + (cb + 1 + (size_t)(i - 1) * 4 + sp) * 512 + h * 64;
        for (int c = 0; c < 64; ++c) op[c] = (h16)o[c];
    }
}

// ---------------- elementwise / glue kernels --------------------------------
__global__ void cvt_f16_kernel(const float* __restrict__ s, h16* __restrict__ d, size_t n)
{
    size_t i = (size_t)blockIdx.x * blockDim.x + threadIdx.x;
    if (i < n) d[i] = (h16)s[i];
}

__global__ void addpos_kernel(float* __restrict__ vid, const float* __restrict__ pos)
{
    size_t i = (size_t)blockIdx.x * blockDim.x + threadIdx.x;
    if (i < (size_t)BATCH * TOK * 128) vid[i] += pos[i % (TOK * 128)];
}

// external (B,1,28) @ ext_w(28,512) + ext_b -> ext (B,512). K=28, tiny: VALU.
__global__ void ext_embed_kernel(const float* __restrict__ e, const float* __restrict__ w,
                                 const float* __restrict__ b, float* __restrict__ o)
{
    const int idx = blockIdx.x * blockDim.x + threadIdx.x;
    if (idx >= BATCH * 512) return;
    const int d = idx % 512, bb = idx / 512;
    float s = b[d];
    for (int j = 0; j < 28; ++j) s += e[bb * 28 + j] * w[j * 512 + d];
    o[idx] = s;
}

// video (B,12,2,32,32) -> patch matrix f16 (B*192, 128)
__global__ void build_patch16_kernel(const float* __restrict__ vdo, h16* __restrict__ o)
{
    const size_t idx = (size_t)blockIdx.x * blockDim.x + threadIdx.x;
    if (idx >= (size_t)BATCH * TOK * 128) return;
    const int feat = (int)(idx % 128);
    const int t    = (int)((idx / 128) % TOK);
    const int b    = (int)(idx / (128 * TOK));
    const int pf = t / 16, s = (t / 4) % 4, n = t % 4;
    const int s1 = s >> 1, s2 = s & 1, n1 = n >> 1, n2 = n & 1;
    const int c = feat / 64, ii = (feat / 8) % 8, jj = feat % 8;
    const int row = s1 * 16 + n1 * 8 + ii;
    const int col = s2 * 16 + n2 * 8 + jj;
    o[idx] = (h16)vdo[(((size_t)b * 12 + pf) * 2 + c) * 1024 + row * 32 + col];
}

__global__ void cat_slot0_kernel(const float* __restrict__ ext, h16* __restrict__ cat16)
{
    const int idx = blockIdx.x * blockDim.x + threadIdx.x;
    if (idx >= BATCH * 512) return;
    const int b = idx / 512, d = idx % 512;
    cat16[(size_t)b * 49 * 512 + d] = (h16)ext[idx];
}

__global__ void build_catres_kernel(const float* __restrict__ ext, const float* __restrict__ x_,
                                    float* __restrict__ dst)
{
    const size_t idx = (size_t)blockIdx.x * blockDim.x + threadIdx.x;
    if (idx >= (size_t)BATCH * 49 * 512) return;
    const int d = (int)(idx % 512);
    const int slot = (int)((idx / 512) % 49);
    const int b = (int)(idx / (49 * 512));
    dst[idx] = (slot == 0) ? ext[b * 512 + d]
                           : x_[((size_t)b * GTOK + slot - 1) * 512 + d];
}

__global__ void upd_ext_kernel(const float* __restrict__ bufO, float* __restrict__ ext)
{
    const int idx = blockIdx.x * blockDim.x + threadIdx.x;
    if (idx >= BATCH * 512) return;
    const int b = idx / 512, d = idx % 512;
    ext[idx] = bufO[(size_t)b * 49 * 512 + d];
}

// vid = x + x_ : x in (B*48,512); x_ pulled from 49-slot bufO slots 1..48
__global__ void upd_vid_kernel(const float* __restrict__ x, const float* __restrict__ bufO,
                               float* __restrict__ vid)
{
    const size_t idx = (size_t)blockIdx.x * blockDim.x + threadIdx.x;
    if (idx >= (size_t)BATCH * GTOK * 512) return;
    const int b = (int)(idx / (GTOK * 512));
    const size_t rem = idx % (GTOK * 512);
    vid[idx] = x[idx] + bufO[(size_t)b * 49 * 512 + 512 + rem];
}

// rows of last frame (pf=11) -> f16 (B*16,128)
__global__ void gather_last16_kernel(const float* __restrict__ vid, h16* __restrict__ o)
{
    const int idx = blockIdx.x * blockDim.x + threadIdx.x;
    if (idx >= BATCH * 16 * 128) return;
    const int d = idx % 128, t = (idx / 128) % 16, b = idx / 2048;
    o[idx] = (h16)vid[((size_t)b * TOK + 176 + t) * 128 + d];
}

// rec (B,16,128) -> out (B,2,32,32)
__global__ void unpatch_kernel(const float* __restrict__ rec, float* __restrict__ out)
{
    const int idx = blockIdx.x * blockDim.x + threadIdx.x;
    if (idx >= BATCH * 2 * 32 * 32) return;
    const int col = idx % 32, row = (idx / 32) % 32;
    const int c = (idx / 1024) % 2, b = idx / 2048;
    const int s1 = row / 16, r16 = row % 16, n1 = r16 / 8, ii = r16 % 8;
    const int s2 = col / 16, c16 = col % 16, n2 = c16 / 8, jj = c16 % 8;
    const int tok  = (s1 * 2 + s2) * 4 + n1 * 2 + n2;
    const int feat = c * 64 + ii * 8 + jj;
    out[idx] = rec[((size_t)b * 16 + tok) * 128 + feat];
}

// ============================================================================
// Host orchestration
// ============================================================================
extern "C" void kernel_launch(void* const* d_in, const int* in_sizes, int n_in,
                              void* d_out, int out_size, void* d_ws, size_t ws_size,
                              hipStream_t stream)
{
    (void)in_sizes; (void)n_in; (void)out_size; (void)ws_size;
    auto F = [&](int i) { return (const float*)d_in[i]; };

    // input indices (setup_inputs flattening order)
    // 0 video, 1 external, 2 patch_w, 3 patch_b, 4 pos, 5 ext_w, 6 ext_b,
    // 7 proj_w, 8 proj_b; then per module (sa=9,sa1=19,ta=29,ta1=39):
    // ln_g, ln_b, wq, bq, wk, bk, wv, bv, wo, bo; ff: 49..54.
    const int SA = 9, SA1 = 19, TA = 29, TA1 = 39, FFB = 49;

    // ---- workspace bump allocator
    char* wp = (char*)d_ws;
    auto alloc = [&](size_t bytes) -> void* {
        void* p = (void*)wp;
        wp += (bytes + 255) & ~(size_t)255;
        return p;
    };

    // f16 weights (converted once per launch; deterministic)
    h16* w_patch = (h16*)alloc((size_t)16384 * 2);
    h16* w_proj  = (h16*)alloc((size_t)16384 * 2);
    h16 *w_sa[4], *w_sa1[4], *w_ta[4], *w_ta1[4];
    for (int j = 0; j < 4; ++j) w_sa[j]  = (h16*)alloc((size_t)8 * 16384 * 2);
    for (int j = 0; j < 4; ++j) w_sa1[j] = (h16*)alloc((size_t)8 * 262144 * 2);
    for (int j = 0; j < 4; ++j) w_ta[j]  = (h16*)alloc((size_t)8 * 16384 * 2);
    for (int j = 0; j < 4; ++j) w_ta1[j] = (h16*)alloc((size_t)8 * 262144 * 2);
    h16* w_ff1 = (h16*)alloc((size_t)8 * 512 * 2048 * 2);
    h16* w_ff2 = (h16*)alloc((size_t)8 * 2048 * 512 * 2);

    // activations
    const size_t M49 = (size_t)BATCH * 49 * 512;   // max token-matrix elems
    float* vid   = (float*)alloc((size_t)BATCH * TOK * 128 * 4);
    float* extb  = (float*)alloc((size_t)BATCH * 512 * 4);
    h16*   ln16  = (h16*)alloc(M49 * 2);
    h16*   at16  = (h16*)alloc(M49 * 2);
    float* qb    = (float*)alloc(M49 * 4);
    float* kb    = (float*)alloc(M49 * 4);
    float* vb    = (float*)alloc(M49 * 4);
    float* bufX  = (float*)alloc((size_t)BATCH * GTOK * 512 * 4);
    float* bufX_ = (float*)alloc((size_t)BATCH * GTOK * 512 * 4);
    float* catr  = (float*)alloc(M49 * 4);
    float* bufO  = (float*)alloc(M49 * 4);
    h16*   hh16  = (h16*)alloc((size_t)BATCH * GTOK * 2048 * 2);
    h16*   eln16 = (h16*)alloc((size_t)BATCH * 512 * 2);
    h16*   eh16  = (h16*)alloc((size_t)BATCH * 2048 * 2);
    h16*   p16   = (h16*)alloc((size_t)BATCH * TOK * 128 * 2);
    h16*   l16   = (h16*)alloc((size_t)BATCH * 16 * 128 * 2);
    float* rec   = (float*)alloc((size_t)BATCH * 16 * 128 * 4);

    // ---- launch helpers
    auto cvt = [&](const float* s, h16* d, size_t n) {
        cvt_f16_kernel<<<dim3((unsigned)((n + 255) / 256)), 256, 0, stream>>>(s, d, n);
    };
    auto gemm = [&](const h16* A, const h16* W, const float* bias, const float* res,
                    float* C, h16* C16, int M, int N, int K, int gelu) {
        dim3 grid((unsigned)(N / BN), (unsigned)(M / BM));
        gemm_wmma_kernel<<<grid, 256, 0, stream>>>(A, W, bias, res, C, C16, M, N, K, gelu);
    };
    auto ln = [&](const float* x, const float* g, const float* b, h16* o,
                  int rows, int dm, int group, int pad) {
        ln_kernel<<<dim3((unsigned)((rows + 7) / 8)), 256, 0, stream>>>(
            x, g, b, o, rows, dm, group, pad);
    };
    auto grid1 = [](size_t n) { return dim3((unsigned)((n + 255) / 256)); };

    // ---- weight conversions (fp32 -> f16), DEPTH folded per tensor
    cvt(F(2), w_patch, 16384);
    cvt(F(7), w_proj, 16384);
    const int wofs[4] = {2, 4, 6, 8};               // wq, wk, wv, wo offsets
    for (int j = 0; j < 4; ++j) {
        cvt(F(SA  + wofs[j]), w_sa[j],  (size_t)8 * 16384);
        cvt(F(SA1 + wofs[j]), w_sa1[j], (size_t)8 * 262144);
        cvt(F(TA  + wofs[j]), w_ta[j],  (size_t)8 * 16384);
        cvt(F(TA1 + wofs[j]), w_ta1[j], (size_t)8 * 262144);
    }
    cvt(F(FFB + 2), w_ff1, (size_t)8 * 512 * 2048);
    cvt(F(FFB + 4), w_ff2, (size_t)8 * 2048 * 512);

    // ---- embeddings
    build_patch16_kernel<<<grid1((size_t)BATCH * TOK * 128), 256, 0, stream>>>(F(0), p16);
    gemm(p16, w_patch, F(3), nullptr, vid, nullptr, BATCH * TOK, 128, 128, 0);
    addpos_kernel<<<grid1((size_t)BATCH * TOK * 128), 256, 0, stream>>>(vid, F(4));
    ext_embed_kernel<<<grid1(BATCH * 512), 256, 0, stream>>>(F(1), F(5), F(6), extb);

    const int M128 = BATCH * TOK;     // 49152
    const int M512 = BATCH * GTOK;    // 12288
    const int M49r = BATCH * 49;      // 12544

    for (int i = 0; i < 8; ++i) {
        // ---- sa: spatial attention over n (dim 128)
        ln(vid, F(SA) + i * 128, F(SA + 1) + i * 128, ln16, M128, 128, 0, 0);
        gemm(ln16, w_sa[0] + (size_t)i * 16384, F(SA + 3) + i * 128, nullptr, qb, nullptr, M128, 128, 128, 0);
        gemm(ln16, w_sa[1] + (size_t)i * 16384, F(SA + 5) + i * 128, nullptr, kb, nullptr, M128, 128, 128, 0);
        gemm(ln16, w_sa[2] + (size_t)i * 16384, F(SA + 7) + i * 128, nullptr, vb, nullptr, M128, 128, 128, 0);
        attn_sa_kernel<<<grid1(BATCH * 4 * 48 * 4), 256, 0, stream>>>(qb, kb, vb, at16);
        gemm(at16, w_sa[3] + (size_t)i * 16384, F(SA + 9) + i * 128, vid, bufX, nullptr, M128, 128, 128, 0);

        // ---- sa1: spatial attention over s (dim 512), x viewed as (B*48,512)
        ln(bufX, F(SA1) + i * 512, F(SA1 + 1) + i * 512, ln16, M512, 512, 0, 0);
        gemm(ln16, w_sa1[0] + (size_t)i * 262144, F(SA1 + 3) + i * 512, nullptr, qb, nullptr, M512, 512, 512, 0);
        gemm(ln16, w_sa1[1] + (size_t)i * 262144, F(SA1 + 5) + i * 512, nullptr, kb, nullptr, M512, 512, 512, 0);
        gemm(ln16, w_sa1[2] + (size_t)i * 262144, F(SA1 + 7) + i * 512, nullptr, vb, nullptr, M512, 512, 512, 0);
        attn_sa1_kernel<<<grid1(BATCH * 8 * 12 * 4), 256, 0, stream>>>(qb, kb, vb, at16);
        gemm(at16, w_sa1[3] + (size_t)i * 262144, F(SA1 + 9) + i * 512, bufX, bufX, nullptr, M512, 512, 512, 0);

        // ---- ta: time attention over f (dim 128), input = layer input vid
        ln(vid, F(TA) + i * 128, F(TA + 1) + i * 128, ln16, M128, 128, 0, 0);
        gemm(ln16, w_ta[0] + (size_t)i * 16384, F(TA + 3) + i * 128, nullptr, qb, nullptr, M128, 128, 128, 0);
        gemm(ln16, w_ta[1] + (size_t)i * 16384, F(TA + 5) + i * 128, nullptr, kb, nullptr, M128, 128, 128, 0);
        gemm(ln16, w_ta[2] + (size_t)i * 16384, F(TA + 7) + i * 128, nullptr, vb, nullptr, M128, 128, 128, 0);
        attn_ta_kernel<<<grid1(BATCH * 4 * 192), 256, 0, stream>>>(qb, kb, vb, at16);
        gemm(at16, w_ta[3] + (size_t)i * 16384, F(TA + 9) + i * 128, vid, bufX_, nullptr, M128, 128, 128, 0);

        // ---- ta1: time attention over (p f) + ext token (dim 512, 49 slots)
        ln(bufX_, F(TA1) + i * 512, F(TA1 + 1) + i * 512, ln16, M512, 512, 48, 1); // slots 1..48
        cat_slot0_kernel<<<grid1(BATCH * 512), 256, 0, stream>>>(extb, ln16);       // slot 0 = ext (no LN)
        gemm(ln16, w_ta1[0] + (size_t)i * 262144, F(TA1 + 3) + i * 512, nullptr, qb, nullptr, M49r, 512, 512, 0);
        gemm(ln16, w_ta1[1] + (size_t)i * 262144, F(TA1 + 5) + i * 512, nullptr, kb, nullptr, M49r, 512, 512, 0);
        gemm(ln16, w_ta1[2] + (size_t)i * 262144, F(TA1 + 7) + i * 512, nullptr, vb, nullptr, M49r, 512, 512, 0);
        attn_ta1_kernel<<<grid1(BATCH * 8 * 4 * 13), 256, 0, stream>>>(qb, kb, vb, at16);
        build_catres_kernel<<<grid1(M49), 256, 0, stream>>>(extb, bufX_, catr);
        gemm(at16, w_ta1[3] + (size_t)i * 262144, F(TA1 + 9) + i * 512, catr, bufO, nullptr, M49r, 512, 512, 0);
        upd_ext_kernel<<<grid1(BATCH * 512), 256, 0, stream>>>(bufO, extb);
        upd_vid_kernel<<<grid1((size_t)BATCH * GTOK * 512), 256, 0, stream>>>(bufX, bufO, vid); // vid = x + x_

        // ---- FF on vid (48 tokens) and ext (1 token), shared weights
        const float* flng = F(FFB) + i * 512;
        const float* flnb = F(FFB + 1) + i * 512;
        const float* fb1  = F(FFB + 3) + i * 2048;
        const float* fb2  = F(FFB + 5) + i * 512;
        h16* f1 = w_ff1 + (size_t)i * 512 * 2048;
        h16* f2 = w_ff2 + (size_t)i * 2048 * 512;
        ln(vid, flng, flnb, ln16, M512, 512, 0, 0);
        gemm(ln16, f1, fb1, nullptr, nullptr, hh16, M512, 2048, 512, 1);   // GELU, f16-only out
        gemm(hh16, f2, fb2, vid, vid, nullptr, M512, 512, 2048, 0);        // + residual
        ln(extb, flng, flnb, eln16, BATCH, 512, 0, 0);
        gemm(eln16, f1, fb1, nullptr, nullptr, eh16, BATCH, 2048, 512, 1);
        gemm(eh16, f2, fb2, extb, extb, nullptr, BATCH, 512, 2048, 0);
    }

    // ---- final projection of last frame + unpatchify
    gather_last16_kernel<<<grid1(BATCH * 16 * 128), 256, 0, stream>>>(vid, l16);
    gemm(l16, w_proj, F(8), nullptr, rec, nullptr, BATCH * 16, 128, 128, 0);
    unpatch_kernel<<<grid1(BATCH * 2 * 32 * 32), 256, 0, stream>>>(rec, (float*)d_out);
}
